// Generator_65386582114398
// MI455X (gfx1250) — matrix operand order repaired and checked by reading, not compiled, not run on hardware
//
#include <hip/hip_runtime.h>
#include <math.h>

typedef __attribute__((ext_vector_type(16))) _Float16 v16h;
typedef __attribute__((ext_vector_type(8)))  _Float16 v8h;
typedef __attribute__((ext_vector_type(8)))  float    v8f;

union Frag16 { v16h v; v8h h[2]; };

#define NBATCH 4096
#define LAT 572
#define LATP 576   // LAT padded to multiple of 32

// =====================================================================
// WMMA GEMM, channel-major activations X (Cin x S), A row-major weights.
// PRECONDITION: Ktot (= Cin, or 3*Cin for CONV) is a multiple of 32.
//   plain: C[m,s] = act( sum_c A[m*lda + c] * X[c*S+s] + b[m] )
//   conv : C[m,s] = act( sum_{dt,c} A[m*lda + c*3 + dt]
//                        * X[c*S + shift_t(s, dt-1)] + b[m] )
// Out-of-range M rows / S columns are CLAMPED (results never stored).
// Rows m in [M, Mpad) are stored as explicit zeros (K-padding support).
// =====================================================================
#define GBM 64
#define GBN 64
#define GBK 32
#define LPITCH 40   // halves per LDS row: 80B, 16B-aligned sub-runs

template<bool CONV>
__global__ __launch_bounds__(256) void gemm_wmma_kernel(
    const float* __restrict__ A, const float* __restrict__ X,
    const float* __restrict__ bias, float* __restrict__ C,
    int M, int Mpad, int Cin, int S, int T, int V, int lda, int actMode)
{
  __shared__ _Float16 As[GBM][LPITCH];   // As[m][k]
  __shared__ _Float16 BsT[GBN][LPITCH];  // BsT[n][k]  (B stored transposed)

  const int tid  = threadIdx.x;
  const int lane = tid & 31;
  const int wave = tid >> 5;
  const int wm   = wave >> 2;            // 0..1 : 32 output rows each
  const int wn   = wave & 3;             // 0..3 : 16 output cols each
  const int mTile = blockIdx.y * GBM;
  const int sTile = blockIdx.x * GBN;
  const int Ktot  = CONV ? 3 * Cin : Cin;

  // staging: thread owns one row (sn) and 8 contiguous k (kg..kg+7)
  const int sn = tid & 63;
  const int kg = (tid >> 6) * 8;         // 0,8,16,24
  const int amc = min(mTile + sn, M - 1);    // clamped A row
  const int gsc = min(sTile + sn, S - 1);    // clamped B column

  // Conv: decompose this thread's spatial column ONCE
  int tcol = 0; size_t bnv = 0;
  if (CONV) {
    const int v   = gsc % V;
    const int rem = gsc / V;
    tcol = rem % T;
    const int n = rem / T;
    bnv = (size_t)n * T * V + v;
  }

  v8f acc0 = {}; v8f acc1 = {};

  for (int k0 = 0; k0 < Ktot; k0 += GBK) {
    // ========= stage A: 8 contiguous k for row amc (no guards) =========
    {
      v8h pack;
      if (!CONV) {
        const float4* p = (const float4*)(A + (size_t)amc * lda + k0 + kg);
        const float4 x0 = p[0], x1 = p[1];
        pack[0] = (_Float16)x0.x; pack[1] = (_Float16)x0.y;
        pack[2] = (_Float16)x0.z; pack[3] = (_Float16)x0.w;
        pack[4] = (_Float16)x1.x; pack[5] = (_Float16)x1.y;
        pack[6] = (_Float16)x1.z; pack[7] = (_Float16)x1.w;
      } else {
#pragma unroll
        for (int j = 0; j < 8; ++j) {
          const int kk = k0 + kg + j;
          const int dt = (kk >= Cin) + (kk >= 2 * Cin);     // no division
          const int c  = kk - dt * Cin;
          pack[j] = (_Float16)A[(size_t)amc * lda + (size_t)c * 3 + dt];
        }
      }
      *(v8h*)&As[sn][kg] = pack;           // one ds_store_b128
    }
    // ========= stage B: 8 contiguous k for column gsc =========
    {
      v8h pack;
      if (!CONV) {
#pragma unroll
        for (int j = 0; j < 8; ++j)        // coalesced across lanes per j
          pack[j] = (_Float16)X[(size_t)(k0 + kg + j) * S + gsc];
        if (k0 + GBK < Ktot)               // gfx1250 global_prefetch_b8
          __builtin_prefetch(X + (size_t)(k0 + GBK + kg) * S + gsc, 0, 1);
      } else {
#pragma unroll
        for (int j = 0; j < 8; ++j) {
          const int kk = k0 + kg + j;
          const int dt = (kk >= Cin) + (kk >= 2 * Cin);
          const int c  = kk - dt * Cin;
          const int ts = tcol + dt - 1;                     // padT = 1
          const bool ok = (ts >= 0) && (ts < T);
          const int tsc = ok ? ts : 0;
          const float v = X[(size_t)c * S + bnv + (size_t)tsc * V];
          pack[j] = ok ? (_Float16)v : (_Float16)0.f;
        }
      }
      *(v8h*)&BsT[sn][kg] = pack;          // one ds_store_b128
    }
    __syncthreads();
    // ---- fragments: 2x ds_load_b128 each, per ISA 7.12.2 layouts ----
    {
      Frag16 a0, a1, b;
      const int mrow = lane & 15;
      const int kh   = lane >> 4;
      a0.h[0] = *(const v8h*)&As[wm * 32 + mrow][kh * 8];
      a0.h[1] = *(const v8h*)&As[wm * 32 + mrow][16 + kh * 8];
      a1.h[0] = *(const v8h*)&As[wm * 32 + 16 + mrow][kh * 8];
      a1.h[1] = *(const v8h*)&As[wm * 32 + 16 + mrow][16 + kh * 8];
      b.h[0]  = *(const v8h*)&BsT[wn * 16 + mrow][kh * 8];
      b.h[1]  = *(const v8h*)&BsT[wn * 16 + mrow][16 + kh * 8];
      acc0 = __builtin_amdgcn_wmma_f32_16x16x32_f16(false, a0.v, false, b.v,
                                                    (short)0, acc0, false, false);
      acc1 = __builtin_amdgcn_wmma_f32_16x16x32_f16(false, a1.v, false, b.v,
                                                    (short)0, acc1, false, false);
    }
    __syncthreads();
  }

  // ---- epilogue: C/D layout = 8 VGPRs, m = r + (lane>=16)*8 ----
  const int s = sTile + wn * 16 + (lane & 15);
  if (s < S) {
#pragma unroll
    for (int r = 0; r < 8; ++r) {
      int m = mTile + wm * 32 + r + ((lane >> 4) << 3);
      if (m < M) {
        float v = acc0[r] + (bias ? bias[m] : 0.f);
        if (actMode == 1) v = (v > 0.f) ? v : 0.2f * v;
        C[(size_t)m * S + s] = v;
      } else if (m < Mpad) {
        C[(size_t)m * S + s] = 0.f;        // zero K-padding rows
      }
      m += 16;
      if (m < M) {
        float v = acc1[r] + (bias ? bias[m] : 0.f);
        if (actMode == 1) v = (v > 0.f) ? v : 0.2f * v;
        C[(size_t)m * S + s] = v;
      } else if (m < Mpad) {
        C[(size_t)m * S + s] = 0.f;
      }
    }
  }
}

// =====================================================================
// Direct small matmul for tiny M/K (block5/6 TCN co=3, block6 GCN K=3).
// One thread per spatial column; X values reused across all M rows.
// NDT = 1 (plain) or 3 (temporal conv, pad 1).  Cin*NDT <= 9, M <= 16.
// =====================================================================
__global__ void small_gemm_kernel(const float* __restrict__ A,
                                  const float* __restrict__ X,
                                  const float* __restrict__ bias,
                                  float* __restrict__ C,
                                  int M, int Cin, int NDT, int S, int T, int V,
                                  int lda)
{
  long s = (long)blockIdx.x * blockDim.x + threadIdx.x;
  if (s >= S) return;
  float xv[9];
  if (NDT == 1) {
    for (int c = 0; c < Cin; ++c) xv[c] = X[(size_t)c * S + s];
  } else {
    const int v   = (int)(s % V);
    const int rem = (int)(s / V);
    const int t   = rem % T;
    const int n   = rem / T;
    const size_t bnv = (size_t)n * T * V + v;
    for (int dt = 0; dt < 3; ++dt) {
      const int ts = t + dt - 1;
      const bool ok = (ts >= 0) && (ts < T);
      const int tsc = ok ? ts : 0;
      for (int c = 0; c < Cin; ++c) {
        const float x = X[(size_t)c * S + bnv + (size_t)tsc * V];
        xv[dt * Cin + c] = ok ? x : 0.f;
      }
    }
  }
  for (int m = 0; m < M; ++m) {
    float acc = bias ? bias[m] : 0.f;
    if (NDT == 1) {
      for (int c = 0; c < Cin; ++c) acc += A[(size_t)m * lda + c] * xv[c];
    } else {
      for (int dt = 0; dt < 3; ++dt)
        for (int c = 0; c < Cin; ++c)
          acc += A[(size_t)m * lda + (size_t)c * 3 + dt] * xv[dt * Cin + c];
    }
    C[(size_t)m * S + s] = acc;
  }
}

// =====================================================================
// Zero-pad weights along K: P[m, k<K] = W[m*K+k], else 0  (Kpad cols)
// =====================================================================
__global__ void padw_kernel(const float* __restrict__ W, float* __restrict__ P,
                            int M, int K, int Kpad)
{
  long gid = (long)blockIdx.x * blockDim.x + threadIdx.x;
  const long total = (long)M * Kpad;
  if (gid >= total) return;
  const int m = (int)(gid / Kpad);
  const int k = (int)(gid % Kpad);
  P[gid] = (k < K) ? W[(size_t)m * K + k] : 0.f;
}

// =====================================================================
// X0 build: channel-major (576, 4096): rows 0..59 emb, 60..571 z, pad 0
// =====================================================================
__global__ void build_x0_kernel(const float* __restrict__ z,
                                const float* __restrict__ emb,
                                const int* __restrict__ labels,
                                float* __restrict__ X)
{
  long gid = (long)blockIdx.x * blockDim.x + threadIdx.x;
  const long total = (long)LATP * NBATCH;
  if (gid >= total) return;
  const int n = (int)(gid % NBATCH);
  const int c = (int)(gid / NBATCH);
  float v = 0.f;
  if (c < 60)        v = emb[labels[n] * 60 + c];
  else if (c < LAT)  v = z[(size_t)n * 512 + (c - 60)];
  X[gid] = v;
}

// =====================================================================
// Upsample-V (U matrices in closed form) + temporal index select, fused.
// Writes Cpad rows; rows c in [C, Cpad) are zero.
// =====================================================================
__device__ inline float Uval(int lvl, int v, int w) {
  if (lvl == 2) return (w == 2) ? 1.f : 0.5f;                 // 1 -> 5
  if (lvl == 1) {                                             // 5 -> 11
    if (w <= 8 && (w & 1) == 0) return (v == (w >> 1)) ? 1.f : 0.f;
    if (w <= 7) return (v == ((w - 1) >> 1) || v == ((w + 1) >> 1)) ? 0.5f : 0.f;
    if (w == 9)  return (v == 4 || v == 0) ? 0.5f : 0.f;
    return (v == 0 || v == 2) ? 0.5f : 0.f;                   // w == 10
  }
  // lvl == 0 : 11 -> 25
  if (w <= 20 && (w & 1) == 0) return (v == (w >> 1)) ? 1.f : 0.f;
  if (w <= 19) return (v == ((w - 1) >> 1) || v == ((w + 1) >> 1)) ? 0.5f : 0.f;
  if (w == 21) return (v == 10 || v == 0) ? 0.5f : 0.f;
  if (w == 22) return (v == 0 || v == 5) ? 0.5f : 0.f;
  if (w == 23) return (v == 1 || v == 6) ? 0.5f : 0.f;
  return (v == 2 || v == 7) ? 0.5f : 0.f;                     // w == 24
}

__global__ void resample_kernel(const float* __restrict__ in, float* __restrict__ out,
                                int C, int Cpad, int Tin, int Vin,
                                int Tout, int Vout, int lvl)
{
  long gid = (long)blockIdx.x * blockDim.x + threadIdx.x;
  const long total = (long)Cpad * NBATCH * Tout * Vout;
  if (gid >= total) return;
  const int w = (int)(gid % Vout);
  long r = gid / Vout;
  const int t = (int)(r % Tout); r /= Tout;
  const int n = (int)(r % NBATCH);
  const int c = (int)(r / NBATCH);
  if (c >= C) { out[gid] = 0.f; return; }
  const int tsrc = (int)(((long)t * Tin) / Tout);
  const float* src = in + (((size_t)c * NBATCH + n) * Tin + tsrc) * Vin;
  float acc;
  if (lvl < 0) {
    acc = src[w];
  } else {
    acc = 0.f;
    for (int v = 0; v < Vin; ++v) acc += Uval(lvl, v, w) * src[v];
  }
  out[gid] = acc;
}

// =====================================================================
// Adjacency mix: Y[o,s] = sum_k sum_v G[k*co+o, (s-w)+v] * A_k[v,w]*ei[k,v,w]
// A = [I, An, An^T], path graph, tri-diagonal sparse.
// =====================================================================
__global__ void amix_kernel(const float* __restrict__ G, const float* __restrict__ ei,
                            float* __restrict__ Y, int co, int S, int V)
{
  long gid = (long)blockIdx.x * blockDim.x + threadIdx.x;
  const long total = (long)co * S;
  if (gid >= total) return;
  const int o = (int)(gid / S);
  const int s = (int)(gid % S);
  if (V == 1) { Y[gid] = G[gid] * ei[0]; return; }   // _adj(1) = ones, K=1
  const int w  = s % V;
  const int sb = s - w;
  float acc = ei[(size_t)w * V + w] * G[(size_t)o * S + sb + w];   // k=0
  const float invw = (w == 0 || w == V - 1) ? 1.f : 0.5f;
#pragma unroll
  for (int dv = -1; dv <= 1; dv += 2) {
    const int v = w + dv;
    if (v >= 0 && v < V) {
      const float invv = (v == 0 || v == V - 1) ? 1.f : 0.5f;
      acc += invv * ei[(size_t)(V + v) * V + w]     * G[(size_t)(co + o)     * S + sb + v];
      acc += invw * ei[(size_t)(2 * V + v) * V + w] * G[(size_t)(2 * co + o) * S + sb + v];
    }
  }
  Y[gid] = acc;
}

// =====================================================================
// Per-channel BN stats: st[2m]=mean, st[2m+1]=rsqrt(var+1e-5)
// =====================================================================
__global__ void stats_kernel(const float* __restrict__ X, float* __restrict__ st, long S)
{
  __shared__ float ssum[256];
  __shared__ float ssq[256];
  const int m = blockIdx.x;
  const float* p = X + (size_t)m * S;
  float s = 0.f, q = 0.f;
  for (long i = threadIdx.x; i < S; i += 256) { const float v = p[i]; s += v; q += v * v; }
  ssum[threadIdx.x] = s; ssq[threadIdx.x] = q;
  __syncthreads();
  for (int off = 128; off > 0; off >>= 1) {
    if ((int)threadIdx.x < off) {
      ssum[threadIdx.x] += ssum[threadIdx.x + off];
      ssq[threadIdx.x]  += ssq[threadIdx.x + off];
    }
    __syncthreads();
  }
  if (threadIdx.x == 0) {
    const float mean = ssum[0] / (float)S;
    const float var  = ssq[0] / (float)S - mean * mean;
    st[2 * m]     = mean;
    st[2 * m + 1] = rsqrtf(var + 1e-5f);
  }
}

// =====================================================================
// Combine: BN(tcn) + BN(res)/identity/none + noise, then act
// =====================================================================
__device__ inline float gauss_noise(unsigned int seed, long s)
{
  unsigned long long x = (((unsigned long long)seed) << 32) ^ (unsigned long long)s;
  x ^= x >> 30; x *= 0xBF58476D1CE4E5B9ull;
  x ^= x >> 27; x *= 0x94D049BB133111EBull;
  x ^= x >> 31;
  const unsigned int u1 = (unsigned int)(x & 0xFFFFFFFFu);
  const unsigned int u2 = (unsigned int)(x >> 32);
  const float f1 = u1 * 2.3283064365e-10f + 1e-12f;
  const float f2 = u2 * 2.3283064365e-10f;
  return sqrtf(-2.f * logf(f1)) * cosf(6.28318530718f * f2);
}

__global__ void combine_kernel(const float* __restrict__ Z, const float* __restrict__ R,
                               const float* __restrict__ Xin,
                               const float* __restrict__ stZ,
                               const float* __restrict__ tg, const float* __restrict__ tb,
                               const float* __restrict__ stR,
                               const float* __restrict__ rg, const float* __restrict__ rb,
                               const float* __restrict__ nw, float* __restrict__ out,
                               int co, long S, int T, int V,
                               int resMode, int useBn, int doTanh, int finalLayout, int seed)
{
  long gid = (long)blockIdx.x * blockDim.x + threadIdx.x;
  const long total = (long)co * S;
  if (gid >= total) return;
  const int  o = (int)(gid / S);
  const long s = gid % S;

  float zv = Z[gid];
  if (useBn) zv = (zv - stZ[2 * o]) * stZ[2 * o + 1] * tg[o] + tb[o];
  float rv = 0.f;
  if (resMode == 1) {
    rv = R[gid];
    rv = (rv - stR[2 * o]) * stR[2 * o + 1] * rg[o] + rb[o];
  } else if (resMode == 2) {
    rv = Xin[gid];
  }
  float y = zv + rv + nw[o] * gauss_noise((unsigned int)seed, s);
  y = doTanh ? tanhf(y) : ((y > 0.f) ? y : 0.2f * y);

  if (finalLayout) {  // (n, o, t, v) to match reference output
    const int v = (int)(s % V);
    long r = s / V;
    const int t = (int)(r % T);
    const int n = (int)(r / T);
    out[((size_t)n * co + o) * ((size_t)T * V) + (size_t)t * V + v] = y;
  } else {
    out[gid] = y;       // channel-major for next block
  }
}

// =====================================================================
// Host orchestration
// =====================================================================
extern "C" void kernel_launch(void* const* d_in, const int* in_sizes, int n_in,
                              void* d_out, int out_size, void* d_ws, size_t ws_size,
                              hipStream_t stream)
{
  (void)in_sizes; (void)n_in; (void)out_size; (void)ws_size;

  const float* z_in  = (const float*)d_in[0];
  const float* emb   = (const float*)d_in[1];
  const float* mlp_w = (const float*)d_in[2];
  const float* mlp_b = (const float*)d_in[3];

  struct BP { const float *gcn_w,*gcn_b,*tcn_w,*tcn_b,*noise_w,*ei,
                          *tcn_g,*tcn_beta,*res_w,*res_b,*res_g,*res_beta; };
  BP bp[7] = {};
  const int CI[7]  = {572,512,256,128,64,32,3};
  const int CO[7]  = {512,256,128,64,32,3,3};
  const int LVL[7] = {3,3,2,2,1,1,0};
  const int BN_[7] = {0,1,0,1,0,1,0};
  const int RES[7] = {0,1,1,1,1,1,2};   // 0 none, 1 conv, 2 identity
  const int UPS[7] = {0,0,1,0,1,0,1};
  const int TT[7]  = {1,4,4,8,16,32,64};
  const int TAN[7] = {0,0,0,0,0,0,1};
  const int VV[7]  = {1,1,5,5,11,11,25};

  int idx = 4;
  for (int i = 0; i < 7; ++i) {
    bp[i].gcn_w   = (const float*)d_in[idx++];
    bp[i].gcn_b   = (const float*)d_in[idx++];
    bp[i].tcn_w   = (const float*)d_in[idx++];
    bp[i].tcn_b   = (const float*)d_in[idx++];
    bp[i].noise_w = (const float*)d_in[idx++];
    bp[i].ei      = (const float*)d_in[idx++];
    if (BN_[i]) { bp[i].tcn_g = (const float*)d_in[idx++];
                  bp[i].tcn_beta = (const float*)d_in[idx++]; }
    if (RES[i] == 1) {
      bp[i].res_w = (const float*)d_in[idx++];
      bp[i].res_b = (const float*)d_in[idx++];
      bp[i].res_g = (const float*)d_in[idx++];
      bp[i].res_beta = (const float*)d_in[idx++];
    }
  }
  const int* labels = (const int*)d_in[idx];

  // Workspace arena (float offsets), lifetime-aware reuse:
  float* ws = (float*)d_ws;
  const size_t offA = 0;                     // block in/out ping (max 23.07M f)
  const size_t offB = offA + 23068672;       // Xin after resample (max 46.14M f)
  const size_t offG = offB + 46137344;       // GCN out (max 69.21M f)
  const size_t offY = offG + 69206016;       // A-mix out (max 23.07M f)
  const size_t offZ = offY + 23068672;       // TCN out
  const size_t offR = offZ + 23068672;       // residual out
  const size_t offSt = offR + 23068672;      // BN stats (2 x 1024 f)
  const size_t offPad = offSt + 2048;        // padded weights (~1.62M f)
  float* bufA = ws + offA;
  float* bufB = ws + offB;
  float* bufG = ws + offG;
  float* bufY = ws + offY;
  float* bufZ = ws + offZ;
  float* bufR = ws + offR;
  float* stZ  = ws + offSt;
  float* stR  = ws + offSt + 1024;
  float* padMlp[4];
  for (int i = 0; i < 4; ++i) padMlp[i] = ws + offPad + (size_t)i * LAT * LATP;
  float* padG0 = ws + offPad + (size_t)4 * LAT * LATP;   // 512 x 576

  auto gemm_plain = [&](const float* A, const float* X, const float* bias, float* C,
                        int M, int Mpad, int Cin, int S, int lda, int act) {
    dim3 g((S + GBN - 1) / GBN, (Mpad + GBM - 1) / GBM);
    gemm_wmma_kernel<false><<<g, 256, 0, stream>>>(A, X, bias, C, M, Mpad, Cin, S,
                                                   1, 1, lda, act);
  };
  auto gemm_conv = [&](const float* A, const float* X, const float* bias, float* C,
                       int M, int Cin, int S, int T, int V, int lda) {
    dim3 g((S + GBN - 1) / GBN, (M + GBM - 1) / GBM);
    gemm_wmma_kernel<true><<<g, 256, 0, stream>>>(A, X, bias, C, M, M, Cin, S,
                                                  T, V, lda, 0);
  };

  // ---- pad the K=572 weights to K=576 (A-side zeros kill B garbage) ----
  {
    const long totM = (long)LAT * LATP;
    for (int i = 0; i < 4; ++i)
      padw_kernel<<<(unsigned)((totM + 255) / 256), 256, 0, stream>>>(
          mlp_w + (size_t)i * LAT * LAT, padMlp[i], LAT, LAT, LATP);
    const long totG = (long)512 * LATP;
    padw_kernel<<<(unsigned)((totG + 255) / 256), 256, 0, stream>>>(
        bp[0].gcn_w, padG0, 512, LAT, LATP);
  }

  // ---- MLP: x = concat(emb[labels], z); 4x leaky_relu(x @ W^T + b) ----
  {
    const long tot = (long)LATP * NBATCH;
    build_x0_kernel<<<(unsigned)((tot + 255) / 256), 256, 0, stream>>>(z_in, emb, labels, bufY);
    float* cur = bufY; float* nxt = bufZ;
    for (int i = 0; i < 4; ++i) {
      float* dst = (i == 3) ? bufA : nxt;
      const int mpad = (i == 3) ? LAT : LATP;   // intermediates carry zero pad rows
      gemm_plain(padMlp[i], cur, mlp_b + i * LAT, dst,
                 LAT, mpad, LATP, NBATCH, LATP, /*leaky*/1);
      float* t = cur; cur = dst; nxt = t;
    }
  }

  // ---- 7 GCN/TCN blocks ----
  int Tprev = 1, Vprev = 1;
  for (int i = 0; i < 7; ++i) {
    const int ci = CI[i], co = CO[i], T = TT[i], V = VV[i];
    const long S = (long)NBATCH * T * V;
    const int K = (V == 1) ? 1 : 3;
    const int cip = (i == 0) ? LATP : ci;      // K-padded input channels

    // resample (identity / V-upsample + T index-select): bufA -> bufB
    {
      const long tot = (long)cip * S;
      resample_kernel<<<(unsigned)((tot + 255) / 256), 256, 0, stream>>>(
          bufA, bufB, ci, cip, Tprev, Vprev, T, V, UPS[i] ? LVL[i] : -1);
    }
    // GCN projection: bufB -> bufG
    if (ci >= 32) {
      const float* gw  = (i == 0) ? padG0 : bp[i].gcn_w;
      gemm_plain(gw, bufB, bp[i].gcn_b, bufG, K * co, K * co, cip, (int)S, cip, 0);
    } else {   // block 6: K=3, direct kernel
      small_gemm_kernel<<<(unsigned)((S + 255) / 256), 256, 0, stream>>>(
          bp[i].gcn_w, bufB, bp[i].gcn_b, bufG, K * co, ci, 1, (int)S, T, V, ci);
    }
    // residual 1x1 conv: bufB -> bufR (all conv-res blocks have ci >= 32)
    if (RES[i] == 1)
      gemm_plain(bp[i].res_w, bufB, bp[i].res_b, bufR, co, co, cip, (int)S, cip, 0);
    // adjacency mix: bufG -> bufY
    {
      const long tot = (long)co * S;
      amix_kernel<<<(unsigned)((tot + 255) / 256), 256, 0, stream>>>(
          bufG, bp[i].ei, bufY, co, (int)S, V);
    }
    // TCN temporal conv: bufY -> bufZ
    if (co >= 32) {
      gemm_conv(bp[i].tcn_w, bufY, bp[i].tcn_b, bufZ, co, co, (int)S, T, V, co * 3);
    } else {   // blocks 5,6: co=3 -> K=9, direct kernel
      small_gemm_kernel<<<(unsigned)((S + 255) / 256), 256, 0, stream>>>(
          bp[i].tcn_w, bufY, bp[i].tcn_b, bufZ, co, co, 3, (int)S, T, V, co * 3);
    }
    // BN stats
    if (BN_[i])      stats_kernel<<<co, 256, 0, stream>>>(bufZ, stZ, S);
    if (RES[i] == 1) stats_kernel<<<co, 256, 0, stream>>>(bufR, stR, S);
    // combine + noise + activation
    {
      float* outp = (i == 6) ? (float*)d_out : bufA;
      const long tot = (long)co * S;
      combine_kernel<<<(unsigned)((tot + 255) / 256), 256, 0, stream>>>(
          bufZ, bufR, bufB, stZ, bp[i].tcn_g, bp[i].tcn_beta,
          stR, bp[i].res_g, bp[i].res_beta, bp[i].noise_w, outp,
          co, S, T, V, RES[i], BN_[i], TAN[i], (i == 6) ? 1 : 0, 1000 + i);
    }
    Tprev = T; Vprev = V;
  }
}
